// fusionEncodeBlock_48782238548122
// MI455X (gfx1250) — matrix operand order repaired and checked by reading, not compiled, not run on hardware
//
#include <hip/hip_runtime.h>

// ---------------------------------------------------------------------------
// fusionEncodeBlock for MI455X (gfx1250): bf16 WMMA GEMMs + VALU band-attention
// GEMM: multi-N-tile waves (A-fragment reuse) -> higher wmma:vmem issue ratio
// ---------------------------------------------------------------------------

typedef __attribute__((ext_vector_type(16))) __bf16 v16bf;
typedef __attribute__((ext_vector_type(8)))  __bf16 v8bf;
typedef __attribute__((ext_vector_type(8)))  float  v8f;

#define DDIM  256
#define SEQ   1024
#define NBAT  2
#define HEADS 4
#define CDIM  64
#define FFHID 2048
#define MROWS 2048           // NBAT * SEQ
#define WINR  10

// --------------------------- weight convert+transpose ----------------------
// in[K][N] f32 (row-major)  ->  out[N][K] bf16 (row-major)  so that GEMM B
// fragments become two contiguous 16B runs per lane.
__global__ void wcvt_transpose(const float* __restrict__ in, __bf16* __restrict__ out,
                               int K, int N)
{
    size_t i = (size_t)blockIdx.x * blockDim.x + threadIdx.x;
    if (i >= (size_t)K * N) return;
    int k = (int)(i / N);
    int n = (int)(i % N);
    out[(size_t)n * K + k] = (__bf16)in[i];
}

// --------------------------- layernorm (D=256), bf16 out -------------------
// One wave per row; 8 channels per lane; wave32 shuffle reductions.
__global__ void layernorm_bf16(const float* __restrict__ x, const float* __restrict__ g,
                               const float* __restrict__ b, __bf16* __restrict__ y)
{
    int wave = threadIdx.x >> 5, lane = threadIdx.x & 31;
    int row  = blockIdx.x * 8 + wave;
    const float* xr = x + (size_t)row * DDIM;
    float v[8]; float s = 0.f;
#pragma unroll
    for (int j = 0; j < 8; ++j) { v[j] = xr[lane + 32 * j]; s += v[j]; }
#pragma unroll
    for (int off = 16; off > 0; off >>= 1) s += __shfl_xor(s, off, 32);
    float mean = s * (1.f / DDIM);
    float var = 0.f;
#pragma unroll
    for (int j = 0; j < 8; ++j) { float d = v[j] - mean; var += d * d; }
#pragma unroll
    for (int off = 16; off > 0; off >>= 1) var += __shfl_xor(var, off, 32);
    float inv = rsqrtf(var * (1.f / DDIM) + 1e-5f);
    __bf16* yr = y + (size_t)row * DDIM;
#pragma unroll
    for (int j = 0; j < 8; ++j) {
        int c = lane + 32 * j;
        yr[c] = (__bf16)((v[j] - mean) * inv * g[c] + b[c]);
    }
}

// --------------------------- WMMA GEMM -------------------------------------
// C[M,N] = act( A[M,K]_bf16 x Bt[N,K]_bf16^T + bias ) (+ res)  -> f32 / bf16
// Each wave computes a 16(M) x NT*16(N) tile: one A fragment is reused for NT
// v_wmma_f32_16x16x32_bf16 issues per K-step (NT=2 for N=256, NT=4 for FF1).
// Fragment packing follows the CDNA5 ISA VGPR layout tables:
//   A (16x32): lane L row M=L&15; half=L>>4: e[0..7]=K[half*8+j], e[8..15]=K[16+half*8+j]
//   B (32x16): lane L col N=L&15; e[0..7]=K[half*8+j], e[8..15]=K[16+half*8+j]
//   D: VGPR r -> row M = r + half*8, col N = L&15
// act: 0=none, 1=relu, 2=sigmoid
template<int NT>
__global__ void gemm_wmma_bf16(const __bf16* __restrict__ A, const __bf16* __restrict__ Bt,
                               const float* __restrict__ bias, const float* __restrict__ res,
                               float* __restrict__ outF, __bf16* __restrict__ outB,
                               int K, int N, int act)
{
    int wave = threadIdx.x >> 5, lane = threadIdx.x & 31;
    int tile = blockIdx.x * 8 + wave;
    int tilesNg = N / (16 * NT);
    int tng = tile % tilesNg;
    int tm  = tile / tilesNg;
    int half = lane >> 4, l15 = lane & 15;

    const __bf16* arow = A + (size_t)(tm * 16 + l15) * K;
    const __bf16* brow[NT];
#pragma unroll
    for (int j = 0; j < NT; ++j)
        brow[j] = Bt + (size_t)(tng * NT * 16 + j * 16 + l15) * K;

    v8f acc[NT] = {};

    for (int k0 = 0; k0 < K; k0 += 32) {
        if (k0 + 128 < K) {              // gfx1250 global_prefetch path
            __builtin_prefetch(arow + k0 + 128, 0, 1);
#pragma unroll
            for (int j = 0; j < NT; ++j) __builtin_prefetch(brow[j] + k0 + 128, 0, 1);
        }
        v8bf alo = *(const v8bf*)(arow + k0 + half * 8);
        v8bf ahi = *(const v8bf*)(arow + k0 + 16 + half * 8);
        v16bf a;
#pragma unroll
        for (int e = 0; e < 8; ++e) { a[e] = alo[e]; a[e + 8] = ahi[e]; }
#pragma unroll
        for (int j = 0; j < NT; ++j) {
            v8bf blo = *(const v8bf*)(brow[j] + k0 + half * 8);
            v8bf bhi = *(const v8bf*)(brow[j] + k0 + 16 + half * 8);
            v16bf b;
#pragma unroll
            for (int e = 0; e < 8; ++e) { b[e] = blo[e]; b[e + 8] = bhi[e]; }
            acc[j] = __builtin_amdgcn_wmma_f32_16x16x32_bf16(
                /*neg_a=*/false, a, /*neg_b=*/false, b,
                /*c_mod=*/(short)0, acc[j], /*reuse_a=*/false, /*reuse_b=*/false);
        }
    }

#pragma unroll
    for (int j = 0; j < NT; ++j) {
        int n = tng * NT * 16 + j * 16 + l15;
        float bv = bias ? bias[n] : 0.f;
#pragma unroll
        for (int r = 0; r < 8; ++r) {
            int m = tm * 16 + half * 8 + r;
            float vv = acc[j][r] + bv;
            if (act == 1)      vv = fmaxf(vv, 0.f);
            else if (act == 2) vv = 1.f / (1.f + __expf(-vv));
            size_t o = (size_t)m * N + n;
            if (res)  vv += res[o];
            if (outF) outF[o] = vv;
            if (outB) outB[o] = (__bf16)vv;
        }
    }
}

// --------------------------- local-window gated attention ------------------
// One wave per (b, s, h). Lane owns channels c=lane and c=lane+32 (C=64).
// scores over t in [s-10, s+10]; softmax; out = (A@V) * gate (gate already
// sigmoid-ed in the G projection GEMM). Writes bf16 for the out-proj GEMM.
__global__ void attn_window(const float* __restrict__ Q, const float* __restrict__ Kb,
                            const float* __restrict__ V, const float* __restrict__ G,
                            const unsigned char* __restrict__ mask, __bf16* __restrict__ out)
{
    int wave = threadIdx.x >> 5, lane = threadIdx.x & 31;
    int idx = blockIdx.x * 8 + wave;       // (b*SEQ + s)*HEADS + h
    int h = idx & (HEADS - 1);
    int m = idx >> 2;                      // b*SEQ + s
    int s = m & (SEQ - 1);
    int b = m >> 10;

    const float* qp = Q + (size_t)m * DDIM + h * CDIM;
    float q0 = qp[lane], q1 = qp[lane + 32];

    float sc[2 * WINR + 1];
    float mx = -3.0e38f;
#pragma unroll
    for (int i = 0; i <= 2 * WINR; ++i) {
        int t = s - WINR + i;
        bool valid = (t >= 0) && (t < SEQ);
        int tc = valid ? t : s;
        const float* kp = Kb + (size_t)(b * SEQ + tc) * DDIM + h * CDIM;
        float p = q0 * kp[lane] + q1 * kp[lane + 32];
#pragma unroll
        for (int off = 16; off > 0; off >>= 1) p += __shfl_xor(p, off, 32);
        p *= 0.125f;                        // 1/sqrt(C), C=64
        if (!valid || mask[b * SEQ + tc]) p = -3.0e38f;
        sc[i] = p;
        mx = fmaxf(mx, p);
    }
    float sum = 0.f;
#pragma unroll
    for (int i = 0; i <= 2 * WINR; ++i) { sc[i] = __expf(sc[i] - mx); sum += sc[i]; }
    float inv = 1.f / sum;

    float o0 = 0.f, o1 = 0.f;
#pragma unroll
    for (int i = 0; i <= 2 * WINR; ++i) {
        int t = s - WINR + i;
        if (t < 0 || t >= SEQ) continue;   // wave-uniform branch
        const float* vp = V + (size_t)(b * SEQ + t) * DDIM + h * CDIM;
        float w = sc[i] * inv;
        o0 += w * vp[lane];
        o1 += w * vp[lane + 32];
    }
    const float* gp = G + (size_t)m * DDIM + h * CDIM;
    __bf16* op = out + (size_t)m * DDIM + h * CDIM;
    op[lane]      = (__bf16)(o0 * gp[lane]);
    op[lane + 32] = (__bf16)(o1 * gp[lane + 32]);
}

// --------------------------- host orchestration ----------------------------
extern "C" void kernel_launch(void* const* d_in, const int* in_sizes, int n_in,
                              void* d_out, int out_size, void* d_ws, size_t ws_size,
                              hipStream_t stream)
{
    (void)in_sizes; (void)n_in; (void)out_size; (void)ws_size;

    const size_t DD  = (size_t)DDIM * DDIM;      // 65536
    const size_t BSD = (size_t)MROWS * DDIM;     // 524288
    const size_t W1  = (size_t)DDIM * FFHID;     // 524288

    const float* inL    = (const float*)d_in[0];
    const float* inN    = (const float*)d_in[1];
    const float* inO    = (const float*)d_in[2];
    const unsigned char* seqMask = (const unsigned char*)d_in[3];
    const float* aWq    = (const float*)d_in[4];
    const float* aWk    = (const float*)d_in[5];
    const float* aWv    = (const float*)d_in[6];
    const float* aGw    = (const float*)d_in[7];
    const float* aGb    = (const float*)d_in[8];
    const float* aOw    = (const float*)d_in[9];
    const float* aOb    = (const float*)d_in[10];
    const float* ln_g   = (const float*)d_in[11];
    const float* ln_b   = (const float*)d_in[12];
    const float* ff_g   = (const float*)d_in[13];
    const float* ff_b   = (const float*)d_in[14];
    const float* ff_w1  = (const float*)d_in[15];
    const float* ff_b1  = (const float*)d_in[16];
    const float* ff_w2  = (const float*)d_in[17];
    const float* ff_b2  = (const float*)d_in[18];

    // L,N,O streams live in-place in d_out (f32).
    float* Lst = (float*)d_out;
    float* Nst = Lst + BSD;
    float* Ost = Nst + BSD;
    hipMemcpyAsync(Lst, inL, BSD * sizeof(float), hipMemcpyDeviceToDevice, stream);
    hipMemcpyAsync(Nst, inN, BSD * sizeof(float), hipMemcpyDeviceToDevice, stream);
    hipMemcpyAsync(Ost, inO, BSD * sizeof(float), hipMemcpyDeviceToDevice, stream);

    // bump-allocate scratch (256B aligned)
    char* wsp = (char*)d_ws;
    auto alloc = [&](size_t bytes) -> void* {
        void* p = (void*)wsp;
        wsp += (bytes + 255) & ~(size_t)255;
        return p;
    };
    __bf16* wqT = (__bf16*)alloc(7 * DD * sizeof(__bf16));
    __bf16* wkT = (__bf16*)alloc(7 * DD * sizeof(__bf16));
    __bf16* wvT = (__bf16*)alloc(7 * DD * sizeof(__bf16));
    __bf16* gwT = (__bf16*)alloc(7 * DD * sizeof(__bf16));
    __bf16* owT = (__bf16*)alloc(7 * DD * sizeof(__bf16));
    __bf16* w1T = (__bf16*)alloc(6 * W1 * sizeof(__bf16));
    __bf16* w2T = (__bf16*)alloc(6 * W1 * sizeof(__bf16));
    __bf16* tLb = (__bf16*)alloc(BSD * sizeof(__bf16));
    __bf16* tNb = (__bf16*)alloc(BSD * sizeof(__bf16));
    __bf16* tOb = (__bf16*)alloc(BSD * sizeof(__bf16));
    __bf16* tFb = (__bf16*)alloc(BSD * sizeof(__bf16));
    __bf16* attb = (__bf16*)alloc(BSD * sizeof(__bf16));
    __bf16* ffh  = (__bf16*)alloc((size_t)MROWS * FFHID * sizeof(__bf16));
    float* Qb  = (float*)alloc(BSD * sizeof(float));
    float* Kb2 = (float*)alloc(BSD * sizeof(float));
    float* Vb  = (float*)alloc(BSD * sizeof(float));
    float* Gb  = (float*)alloc(BSD * sizeof(float));

    // convert + transpose all weights to bf16 [N][K] (L2-resident afterwards)
    auto T = [&](const float* in, __bf16* out, int K, int N) {
        size_t tot = (size_t)K * N;
        wcvt_transpose<<<(unsigned)((tot + 255) / 256), 256, 0, stream>>>(in, out, K, N);
    };
    for (int k = 0; k < 7; ++k) {
        T(aWq + k * DD, wqT + k * DD, DDIM, DDIM);
        T(aWk + k * DD, wkT + k * DD, DDIM, DDIM);
        T(aWv + k * DD, wvT + k * DD, DDIM, DDIM);
        T(aGw + k * DD, gwT + k * DD, DDIM, DDIM);
        T(aOw + k * DD, owT + k * DD, DDIM, DDIM);
    }
    for (int k = 0; k < 6; ++k) {
        T(ff_w1 + k * W1, w1T + k * W1, DDIM, FFHID);
        T(ff_w2 + k * W1, w2T + k * W1, FFHID, DDIM);
    }

    auto GEMM = [&](const __bf16* A, const __bf16* Bt, const float* bias, const float* res,
                    float* oF, __bf16* oB, int K, int N, int act) {
        if (N >= 1024) {   // wide GEMM (FF1): 16x64 tiles per wave
            int blocks = (MROWS / 16) * (N / 64) / 8;
            gemm_wmma_bf16<4><<<blocks, 256, 0, stream>>>(A, Bt, bias, res, oF, oB, K, N, act);
        } else {           // N=256 GEMMs: 16x32 tiles keep 128 blocks in flight
            int blocks = (MROWS / 16) * (N / 32) / 8;
            gemm_wmma_bf16<2><<<blocks, 256, 0, stream>>>(A, Bt, bias, res, oF, oB, K, N, act);
        }
    };
    auto LN = [&](const float* x, const float* g, const float* b, __bf16* out) {
        layernorm_bf16<<<MROWS / 8, 256, 0, stream>>>(x, g, b, out);
    };
    auto ATT = [&](int k, const __bf16* xq, const __bf16* xk, const __bf16* xv, float* st) {
        GEMM(xq, wqT + k * DD, nullptr, nullptr, Qb, nullptr, DDIM, DDIM, 0);
        GEMM(xk, wkT + k * DD, nullptr, nullptr, Kb2, nullptr, DDIM, DDIM, 0);
        GEMM(xv, wvT + k * DD, nullptr, nullptr, Vb, nullptr, DDIM, DDIM, 0);
        GEMM(xv, gwT + k * DD, aGb + k * DDIM, nullptr, Gb, nullptr, DDIM, DDIM, 2);
        attn_window<<<MROWS * HEADS / 8, 256, 0, stream>>>(Qb, Kb2, Vb, Gb, seqMask, attb);
        GEMM(attb, owT + k * DD, aOb + k * DDIM, st, st, nullptr, DDIM, DDIM, 0);
    };
    auto FF = [&](int k, float* x) {
        LN(x, ff_g + k * DDIM, ff_b + k * DDIM, tFb);
        GEMM(tFb, w1T + k * W1, ff_b1 + (size_t)k * FFHID, nullptr, nullptr, ffh, DDIM, FFHID, 1);
        GEMM(ffh, w2T + k * W1, ff_b2 + k * DDIM, x, x, nullptr, FFHID, DDIM, 0);
    };

    for (int it = 0; it < 2; ++it) {
        LN(Lst, ln_g + 0 * DDIM, ln_b + 0 * DDIM, tLb); ATT(0, tLb, tLb, tLb, Lst); FF(0, Lst);
        LN(Nst, ln_g + 1 * DDIM, ln_b + 1 * DDIM, tNb); ATT(1, tNb, tNb, tNb, Nst); FF(1, Nst);
        LN(Ost, ln_g + 2 * DDIM, ln_b + 2 * DDIM, tOb); ATT(2, tOb, tOb, tOb, Ost); FF(2, Ost);
        LN(Ost, ln_g + 3 * DDIM, ln_b + 3 * DDIM, tOb);
        LN(Lst, ln_g + 4 * DDIM, ln_b + 4 * DDIM, tLb); ATT(3, tLb, tOb, tOb, Lst); FF(3, Lst);
        LN(Nst, ln_g + 5 * DDIM, ln_b + 5 * DDIM, tNb); ATT(4, tNb, tOb, tOb, Nst); FF(4, Nst);
        LN(Lst, ln_g + 6 * DDIM, ln_b + 6 * DDIM, tLb);
        LN(Nst, ln_g + 7 * DDIM, ln_b + 7 * DDIM, tNb);
        LN(Ost, ln_g + 8 * DDIM, ln_b + 8 * DDIM, tOb);
        ATT(5, tOb, tNb, tNb, Ost);
        ATT(6, tOb, tLb, tLb, Ost);
        FF(5, Ost);
    }
}
